// WindowAttention_2680059592803
// MI455X (gfx1250) — matrix operand order repaired and checked by reading, not compiled, not run on hardware
//
#include <hip/hip_runtime.h>
#include <hip/hip_bf16.h>
#include <math.h>

typedef __attribute__((ext_vector_type(16))) _Float16 v16h;
typedef __attribute__((ext_vector_type(8)))  float    v8f;

#define HEADS   6
#define HSP     8
#define WSP     8
#define NTOK    64          // tokens per window
#define HD      32          // head dim
#define IMG     128
#define NWIN    256         // windows per image
#define CCH     192
#define PD      12
#define LN_EPS  1e-5f

// ---------------------------------------------------------------------------
// Kernel 1: DynamicPosBias MLP over 225 relative offsets -> pos[225][6]
// ---------------------------------------------------------------------------
__global__ void pos_mlp_kernel(const float* __restrict__ rpe,
                               const float* __restrict__ w0, const float* __restrict__ b0,
                               const float* __restrict__ g1, const float* __restrict__ be1,
                               const float* __restrict__ w1, const float* __restrict__ bb1,
                               const float* __restrict__ g2, const float* __restrict__ be2,
                               const float* __restrict__ w2, const float* __restrict__ bb2,
                               const float* __restrict__ g3, const float* __restrict__ be3,
                               const float* __restrict__ w3, const float* __restrict__ bb3,
                               float* __restrict__ pos_out) {
    int i = blockIdx.x * blockDim.x + threadIdx.x;
    if (i >= 225) return;

    float x0 = rpe[2 * i + 0];
    float x1 = rpe[2 * i + 1];

    float h[PD], t[PD];
    #pragma unroll
    for (int j = 0; j < PD; ++j)
        h[j] = x0 * w0[0 * PD + j] + x1 * w0[1 * PD + j] + b0[j];

    const float* gs[3]  = {g1, g2, g3};
    const float* bes[3] = {be1, be2, be3};
    const float* ws[3]  = {w1, w2, w3};
    const float* bbs[3] = {bb1, bb2, bb3};

    #pragma unroll
    for (int r = 0; r < 3; ++r) {
        float m = 0.f;
        #pragma unroll
        for (int j = 0; j < PD; ++j) m += h[j];
        m *= (1.0f / PD);
        float v = 0.f;
        #pragma unroll
        for (int j = 0; j < PD; ++j) { float d = h[j] - m; v += d * d; }
        v *= (1.0f / PD);
        float inv = rsqrtf(v + LN_EPS);
        #pragma unroll
        for (int j = 0; j < PD; ++j) {
            float a = (h[j] - m) * inv * gs[r][j] + bes[r][j];
            t[j] = a > 0.f ? a : 0.f;            // ReLU
        }
        int nout = (r == 2) ? HEADS : PD;
        for (int j = 0; j < nout; ++j) {
            float acc = bbs[r][j];
            #pragma unroll
            for (int k = 0; k < PD; ++k) acc += t[k] * ws[r][k * nout + j];
            h[j] = acc;
        }
    }
    #pragma unroll
    for (int hh = 0; hh < HEADS; ++hh) pos_out[i * HEADS + hh] = h[hh];
}

// ---------------------------------------------------------------------------
// Kernel 2: expand pos table through rel_idx -> bias[6][64][64]
// ---------------------------------------------------------------------------
__global__ void bias_expand_kernel(const float* __restrict__ pos,
                                   const int* __restrict__ rel_idx,
                                   float* __restrict__ bias) {
    int e = blockIdx.x * blockDim.x + threadIdx.x;     // 0 .. 6*4096
    if (e >= HEADS * NTOK * NTOK) return;
    int h = e / (NTOK * NTOK);
    int i = e - h * (NTOK * NTOK);
    bias[e] = pos[rel_idx[i] * HEADS + h];
}

// ---------------------------------------------------------------------------
// Kernel 3: window attention, one block per (window, head), 128 threads.
// WMMA fragment conventions (CDNA5 ISA 7.12.2, wave32):
//   A 16x32 f16 : lane L -> row = L&15, halves[0..15] = K (L>>4)*16 .. +15 (contiguous k order
//                 chosen here, not the ISA table order)
//   B 32x16 f16 : lane L -> col = L&15, halves[0..15] = K (L>>4)*16 .. +15 (same k order as A)
//   C/D 16x16 f32: lane L -> col = L&15, VGPR r -> row r + (L>>4)*8
// ---------------------------------------------------------------------------
__global__ __launch_bounds__(128)
void window_attn_kernel(const float* __restrict__ qkv,
                        const float* __restrict__ bias_tab,
                        float* __restrict__ out) {
    const int h    = blockIdx.x;          // head 0..5
    const int w    = blockIdx.y;          // window 0..2047
    const int b    = w >> 8;              // image
    const int wi   = w & 255;
    const int wy   = wi >> 4;
    const int wx   = wi & 15;
    const int tid  = threadIdx.x;
    const int lane = tid & 31;
    const int wave = tid >> 5;            // 0..3 -> 16-row block

    __shared__ _Float16 Qs[NTOK * HD];        // [t][d], pre-scaled
    __shared__ _Float16 Ks[NTOK * HD];        // [t][d]
    __shared__ _Float16 Vt[HD * NTOK];        // [d][t]  (transposed)
    __shared__ float    Bias[NTOK * NTOK];    // [row][col]
    __shared__ float    Attn[NTOK * NTOK];    // [row][col]; reused as OutS
    __shared__ _Float16 Probs[NTOK * NTOK];   // [row][col]

    const float scale = 0.17677669529663687f;             // 32^-0.5
    const size_t plane = (size_t)8 * 16384 * CCH;         // qkv[0/1/2] stride

    // ---- stage Q/K/V tiles into LDS (f16) with b128 global loads ----
    #pragma unroll
    for (int e = tid; e < (NTOK * HD) / 4; e += 128) {    // 512 float4s
        int t  = e >> 3;
        int d4 = (e & 7) * 4;
        int y  = wy * HSP + (t >> 3);
        int x  = wx * WSP + (t & 7);
        size_t base = (((size_t)b * (IMG * IMG)) + (size_t)y * IMG + x) * CCH
                    + h * HD + d4;
        float4 qv = *(const float4*)(qkv + base);
        float4 kv = *(const float4*)(qkv + plane + base);
        float4 vv = *(const float4*)(qkv + 2 * plane + base);

        _Float16* qd = &Qs[t * HD + d4];
        qd[0] = (_Float16)(qv.x * scale); qd[1] = (_Float16)(qv.y * scale);
        qd[2] = (_Float16)(qv.z * scale); qd[3] = (_Float16)(qv.w * scale);

        _Float16* kd = &Ks[t * HD + d4];
        kd[0] = (_Float16)kv.x; kd[1] = (_Float16)kv.y;
        kd[2] = (_Float16)kv.z; kd[3] = (_Float16)kv.w;

        Vt[(d4 + 0) * NTOK + t] = (_Float16)vv.x;
        Vt[(d4 + 1) * NTOK + t] = (_Float16)vv.y;
        Vt[(d4 + 2) * NTOK + t] = (_Float16)vv.z;
        Vt[(d4 + 3) * NTOK + t] = (_Float16)vv.w;
    }
    {
        const float4* bsrc = (const float4*)(bias_tab + h * (NTOK * NTOK));
        float4*       bdst = (float4*)Bias;
        #pragma unroll
        for (int e = tid; e < (NTOK * NTOK) / 4; e += 128)
            bdst[e] = bsrc[e];
    }
    __syncthreads();

    // ---- phase 1: Attn = Q * K^T + bias (per-wave 16-row block) ----
    {
        const int m0   = wave * 16;
        const int mloc = lane & 15;
        const int kb   = (lane >> 4) * 16;   // A-fragment K base
        const int n    = lane & 15;          // B-fragment column
        const int kb2  = (lane >> 4) * 16;   // B-fragment K base

        v16h a;
        const _Float16* pa = &Qs[(m0 + mloc) * HD + kb];
        #pragma unroll
        for (int i = 0; i < 8; ++i) { a[i] = pa[i]; a[8 + i] = pa[8 + i]; }

        #pragma unroll
        for (int j = 0; j < 4; ++j) {
            v16h bf;
            const _Float16* pb = &Ks[(j * 16 + n) * HD + kb2];
            #pragma unroll
            for (int i = 0; i < 16; ++i) bf[i] = pb[i];

            v8f c = {};
            c = __builtin_amdgcn_wmma_f32_16x16x32_f16(
                    false, a, false, bf, (short)0, c, false, false);

            const int col   = j * 16 + (lane & 15);
            const int rbase = m0 + ((lane >> 4) * 8);
            #pragma unroll
            for (int r = 0; r < 8; ++r) {
                int row = rbase + r;
                Attn[row * NTOK + col] = c[r] + Bias[row * NTOK + col];
            }
        }
    }
    __syncthreads();

    // ---- softmax over rows (one thread per row) ----
    if (tid < NTOK) {
        float mx = -3.4e38f;
        for (int c = 0; c < NTOK; ++c) mx = fmaxf(mx, Attn[tid * NTOK + c]);
        float s = 0.f;
        for (int c = 0; c < NTOK; ++c) {
            float e = __expf(Attn[tid * NTOK + c] - mx);
            s += e;
            Attn[tid * NTOK + c] = e;
        }
        float inv = 1.0f / s;
        for (int c = 0; c < NTOK; ++c)
            Probs[tid * NTOK + c] = (_Float16)(Attn[tid * NTOK + c] * inv);
    }
    __syncthreads();
    // Attn region is dead from here on -> reuse as the output staging buffer
    float* OutS = Attn;                       // [t][c_local] 64 x 32 f32

    // ---- phase 2: Out = Probs * V (K-dim 64 = 2 chained WMMAs) ----
    {
        const int m0   = wave * 16;
        const int mloc = lane & 15;
        const int kb   = (lane >> 4) * 16;
        const int n    = lane & 15;
        const int kb2  = (lane >> 4) * 16;

        #pragma unroll
        for (int j = 0; j < 2; ++j) {        // 16-channel column tiles
            v8f acc = {};
            #pragma unroll
            for (int kc = 0; kc < 2; ++kc) { // token chunks of 32
                v16h a, bf;
                const _Float16* pa = &Probs[(m0 + mloc) * NTOK + kc * 32 + kb];
                #pragma unroll
                for (int i = 0; i < 8; ++i) { a[i] = pa[i]; a[8 + i] = pa[8 + i]; }
                const _Float16* pb = &Vt[(j * 16 + n) * NTOK + kc * 32 + kb2];
                #pragma unroll
                for (int i = 0; i < 16; ++i) bf[i] = pb[i];
                acc = __builtin_amdgcn_wmma_f32_16x16x32_f16(
                        false, a, false, bf, (short)0, acc, false, false);
            }
            // stage D fragment into LDS (row = token, col = local channel)
            const int cl    = j * 16 + (lane & 15);
            const int rbase = m0 + ((lane >> 4) * 8);
            #pragma unroll
            for (int r = 0; r < 8; ++r)
                OutS[(rbase + r) * HD + cl] = acc[r];
        }
    }
    __syncthreads();

    // ---- coalesced b128 stores to (B, H, W, C) ----
    {
        const float4* osrc = (const float4*)OutS;
        #pragma unroll
        for (int e = tid; e < (NTOK * HD) / 4; e += 128) {  // 512 float4s
            int t  = e >> 3;
            int c4 = (e & 7) * 4;
            int y  = wy * HSP + (t >> 3);
            int x  = wx * WSP + (t & 7);
            size_t o = (((size_t)b * IMG + y) * IMG + x) * CCH + h * HD + c4;
            *(float4*)(out + o) = osrc[e];
        }
    }
}

// ---------------------------------------------------------------------------
extern "C" void kernel_launch(void* const* d_in, const int* in_sizes, int n_in,
                              void* d_out, int out_size, void* d_ws, size_t ws_size,
                              hipStream_t stream) {
    const float* qkv     = (const float*)d_in[0];
    const float* rpe     = (const float*)d_in[1];
    const float* w0      = (const float*)d_in[2];
    const float* b0      = (const float*)d_in[3];
    const float* g1      = (const float*)d_in[4];
    const float* be1     = (const float*)d_in[5];
    const float* w1      = (const float*)d_in[6];
    const float* bb1     = (const float*)d_in[7];
    const float* g2      = (const float*)d_in[8];
    const float* be2     = (const float*)d_in[9];
    const float* w2      = (const float*)d_in[10];
    const float* bb2     = (const float*)d_in[11];
    const float* g3      = (const float*)d_in[12];
    const float* be3     = (const float*)d_in[13];
    const float* w3      = (const float*)d_in[14];
    const float* bb3     = (const float*)d_in[15];
    const int*   rel_idx = (const int*)d_in[16];
    float*       out     = (float*)d_out;

    // workspace: pos table (225*6 f32) then bias table (6*64*64 f32)
    float* pos_ws  = (float*)d_ws;
    float* bias_ws = (float*)((char*)d_ws + ((225 * HEADS * sizeof(float) + 255) & ~255));

    pos_mlp_kernel<<<1, 256, 0, stream>>>(rpe, w0, b0, g1, be1, w1, bb1,
                                          g2, be2, w2, bb2, g3, be3, w3, bb3,
                                          pos_ws);

    int nb = (HEADS * NTOK * NTOK + 255) / 256;
    bias_expand_kernel<<<nb, 256, 0, stream>>>(pos_ws, rel_idx, bias_ws);

    dim3 grid(HEADS, 8 * NWIN);   // (head, window) = (6, 2048)
    window_attn_kernel<<<grid, 128, 0, stream>>>(qkv, bias_ws, out);
}
